// BiSSMBlock_21852793602787
// MI455X (gfx1250) — compile-verified, matrix operand order checked by reading, not run on hardware
//
#include <hip/hip_runtime.h>
#include <hip/hip_bf16.h>
#include <cstdint>
#include <cstddef>

// ---------------------------------------------------------------------------
// BiSSM (bidirectional Mamba) block for MI455X / gfx1250.
//   D_MODEL=1024, D_INNER=2048, D_STATE=16, D_CONV=4, DT_RANK=64, B=2, L=2048
// GEMMs run as bf16 WMMA (v_wmma_f32_16x16x32_bf16) with fp32 accumulate and
// register double-buffered fragment loads.
// ---------------------------------------------------------------------------

#define NBATCH  2
#define LSEQ    2048
#define TOKS    (NBATCH * LSEQ)   // 4096
#define DMODEL  1024
#define DINNER  2048
#define NSTATE  16
#define DTRANK  64

typedef __attribute__((ext_vector_type(16))) __bf16 bf16x16;
typedef __attribute__((ext_vector_type(8)))  float  f32x8;

union FragB16 { bf16x16 v; uint4 q[2]; };

// ------------------------------ epilogues ----------------------------------

struct EpiBF16 {                    // store bf16 with column offset
  __bf16* C; int ldc; int coloff;
  __device__ void operator()(int m, int n, float v) const {
    C[(size_t)m * ldc + coloff + n] = (__bf16)v;
  }
};

struct EpiDbc {                     // split [dt(64) | B(16) | C(16)]
  __bf16* dt; float* Bm; float* Cm;
  __device__ void operator()(int m, int n, float v) const {
    if (n < DTRANK)               dt[(size_t)m * DTRANK + n] = (__bf16)v;
    else if (n < DTRANK + NSTATE) Bm[(size_t)m * NSTATE + (n - DTRANK)] = v;
    else                          Cm[(size_t)m * NSTATE + (n - DTRANK - NSTATE)] = v;
  }
};

struct EpiDelta {                   // softplus(acc + dt_b) -> bf16
  const float* bias; __bf16* out;
  __device__ void operator()(int m, int n, float v) const {
    float x  = v + bias[n];
    float sp = (x > 20.f) ? x : log1pf(__expf(x));
    out[(size_t)m * DINNER + n] = (__bf16)sp;
  }
};

struct EpiCombine {                 // acc + bias + residual -> f32 output
  const float* bias; const float* resid; float* out;
  __device__ void operator()(int m, int n, float v) const {
    out[(size_t)m * DMODEL + n] = v + bias[n] + resid[(size_t)m * DMODEL + n];
  }
};

// ------------------------------ WMMA GEMM ----------------------------------
// C[M,N] = A[M,K] @ W[N,K]^T.  A,W bf16 row-major (K contiguous) -> both the
// A-matrix layout (lane m=lane&15, 8 contiguous K at 8*hi and 16+8*hi) and
// the B-matrix layout (lane n=lane&15, same K pattern down column n of B ==
// row n of W) are satisfied by plain 16-byte global loads.
// Workgroup = 8 waves (4 M x 2 N); each wave computes a 32x64 block as
// 2x4 WMMA 16x16 tiles, reusing A frags over N and B frags over M.
// Fragments are double-buffered in registers (compile-time buffer index) so
// the next k-step's 12 b128 loads issue before the current step's 8 WMMAs.
//
// GUARD=false: N is a multiple of the workgroup tile -> branch-free K loop.
// GUARD=true : clamp out-of-range fragment rows to 0 (their products only
//              reach output columns >= N, whose stores are guarded).

template <bool GUARD, typename Epi>
__global__ __launch_bounds__(256) void gemm_bf16_wmma(
    const __bf16* __restrict__ A, int lda,
    const __bf16* __restrict__ W, int ldw,
    int N, int K, Epi epi)
{
  const int lane = threadIdx.x & 31;
  const int wave = threadIdx.x >> 5;
  const int hi   = lane >> 4;      // half-wave selector (K sub-block)
  const int lo   = lane & 15;      // row (A) / column (B,C)
  const int m0 = blockIdx.x * 128 + (wave & 3) * 32;
  const int n0 = blockIdx.y * 128 + (wave >> 2) * 64;

  // Loop-invariant 32-bit element offsets (all buffers << 4 GiB).
  uint32_t aoff[2], boff[4];
#pragma unroll
  for (int mi = 0; mi < 2; ++mi)
    aoff[mi] = (uint32_t)(m0 + mi * 16 + lo) * (uint32_t)lda + 8u * (uint32_t)hi;
#pragma unroll
  for (int ni = 0; ni < 4; ++ni) {
    int n = n0 + ni * 16 + lo;
    if (GUARD && n >= N) n = 0;           // clamp; epilogue guards the stores
    boff[ni] = (uint32_t)n * (uint32_t)ldw + 8u * (uint32_t)hi;
  }

  f32x8 acc[2][4];
#pragma unroll
  for (int mi = 0; mi < 2; ++mi)
#pragma unroll
    for (int ni = 0; ni < 4; ++ni)
#pragma unroll
      for (int r = 0; r < 8; ++r) acc[mi][ni][r] = 0.f;

  FragB16 aF[2][2], bF[2][4];           // [buffer][fragment]

#define LOAD_FRAGS(BUF, KK)                                                  \
  do {                                                                       \
    _Pragma("unroll")                                                        \
    for (int mi = 0; mi < 2; ++mi) {                                         \
      const __bf16* p = A + aoff[mi] + (KK);                                 \
      aF[BUF][mi].q[0] = *(const uint4*)p;                                   \
      aF[BUF][mi].q[1] = *(const uint4*)(p + 16);                            \
    }                                                                        \
    _Pragma("unroll")                                                        \
    for (int ni = 0; ni < 4; ++ni) {                                         \
      const __bf16* p = W + boff[ni] + (KK);                                 \
      bF[BUF][ni].q[0] = *(const uint4*)p;                                   \
      bF[BUF][ni].q[1] = *(const uint4*)(p + 16);                            \
      __builtin_prefetch(p + 64, 0, 1);  /* global_prefetch_b8, k+64 ahead */\
    }                                                                        \
  } while (0)

#define MMA_BLOCK(BUF)                                                       \
  do {                                                                       \
    _Pragma("unroll")                                                        \
    for (int mi = 0; mi < 2; ++mi)                                           \
      _Pragma("unroll")                                                      \
      for (int ni = 0; ni < 4; ++ni)                                         \
        acc[mi][ni] = __builtin_amdgcn_wmma_f32_16x16x32_bf16(               \
            false, aF[BUF][mi].v, false, bF[BUF][ni].v, (short)0,            \
            acc[mi][ni], false, false);                                      \
  } while (0)

  // Software pipeline: buffer 0 always holds frags for step k0 at loop head.
  LOAD_FRAGS(0, 0u);
  uint32_t k0 = 0;
  for (; k0 + 64 < (uint32_t)K; k0 += 64) {
    LOAD_FRAGS(1, k0 + 32);
    MMA_BLOCK(0);
    LOAD_FRAGS(0, k0 + 64);
    MMA_BLOCK(1);
  }
  if (k0 + 32 < (uint32_t)K) {          // two steps remain
    LOAD_FRAGS(1, k0 + 32);
    MMA_BLOCK(0);
    MMA_BLOCK(1);
  } else {                              // one step remains
    MMA_BLOCK(0);
  }

#undef LOAD_FRAGS
#undef MMA_BLOCK

  // C/D layout: lane lo = column, VGPR r holds row r + 8*hi of the 16x16 tile
#pragma unroll
  for (int mi = 0; mi < 2; ++mi)
#pragma unroll
    for (int ni = 0; ni < 4; ++ni) {
      const int n = n0 + ni * 16 + lo;
      if (GUARD && n >= N) continue;
#pragma unroll
      for (int r = 0; r < 8; ++r)
        epi(m0 + mi * 16 + 8 * hi + r, n, acc[mi][ni][r]);
    }
}

// ------------------------------ elementwise --------------------------------

__global__ __launch_bounds__(256) void f32_to_bf16_kernel(
    const float* __restrict__ s, __bf16* __restrict__ d, int n)
{
  int i = blockIdx.x * 256 + threadIdx.x;
  if (i < n) d[i] = (__bf16)s[i];
}

__global__ __launch_bounds__(256) void layernorm_bf16_kernel(
    const float* __restrict__ x, const float* __restrict__ g,
    const float* __restrict__ bta, __bf16* __restrict__ xn)
{
  __shared__ float s1[256], s2[256];
  const int tok = blockIdx.x;
  const float* row = x + (size_t)tok * DMODEL;
  float v[4], a = 0.f, b2 = 0.f;
#pragma unroll
  for (int i = 0; i < 4; ++i) {
    v[i] = row[threadIdx.x + 256 * i];
    a += v[i]; b2 += v[i] * v[i];
  }
  s1[threadIdx.x] = a; s2[threadIdx.x] = b2;
  __syncthreads();
  for (int st = 128; st > 0; st >>= 1) {
    if ((int)threadIdx.x < st) {
      s1[threadIdx.x] += s1[threadIdx.x + st];
      s2[threadIdx.x] += s2[threadIdx.x + st];
    }
    __syncthreads();
  }
  const float mu  = s1[0] * (1.f / DMODEL);
  const float var = s2[0] * (1.f / DMODEL) - mu * mu;
  const float inv = rsqrtf(var + 1e-5f);
#pragma unroll
  for (int i = 0; i < 4; ++i) {
    const int c = threadIdx.x + 256 * i;
    xn[(size_t)tok * DMODEL + c] = (__bf16)((v[i] - mu) * inv * g[c] + bta[c]);
  }
}

// Depthwise causal (forward) / anti-causal (backward) conv + SiLU.
// Reading xi from the in-proj output xz (columns [0, DINNER)).
__global__ __launch_bounds__(256) void conv_silu_kernel(
    const __bf16* __restrict__ xz,    // [TOKS, 2*DINNER]
    const float*  __restrict__ cw,    // [DINNER, 4]
    const float*  __restrict__ cb,    // [DINNER]
    __bf16* __restrict__ u,           // [TOKS, DINNER]
    int reverse)
{
  const int idx = blockIdx.x * 256 + threadIdx.x;   // TOKS*DINNER threads
  const int d   = idx & (DINNER - 1);
  const int tok = idx >> 11;                        // / DINNER
  const int bb  = tok >> 11;                        // / LSEQ
  const int t   = tok & (LSEQ - 1);
  float acc = cb[d];
#pragma unroll
  for (int k = 0; k < 4; ++k) {
    const int tt = reverse ? (t + 3 - k) : (t - 3 + k);
    if (tt >= 0 && tt < LSEQ)
      acc += cw[d * 4 + k] *
             (float)xz[(size_t)(bb * LSEQ + tt) * (2 * DINNER) + d];
  }
  const float s = acc / (1.f + __expf(-acc));       // silu
  u[(size_t)tok * DINNER + d] = (__bf16)s;
}

// Selective scan: one thread per (batch, channel). h[16] lives in VGPRs.
// Forward iterates t ascending; backward descending (time-space formulation,
// so no data reversal anywhere). Fuses the D-skip and silu(z) gate, and
// writes y in place over the delta buffer (each element read before written
// by the same thread -- note: delta/y deliberately NOT __restrict__).
__global__ __launch_bounds__(256) void ssm_scan_kernel(
    const __bf16* delta,               // [TOKS, DINNER] (aliases y)
    const __bf16* __restrict__ u,      // [TOKS, DINNER]
    const float*  __restrict__ Bm,     // [TOKS, NSTATE]
    const float*  __restrict__ Cm,     // [TOKS, NSTATE]
    const float*  __restrict__ A_log,  // [DINNER, NSTATE]
    const float*  __restrict__ Dp,     // [DINNER]
    const __bf16* __restrict__ xz,     // [TOKS, 2*DINNER] (z = cols DINNER..)
    __bf16* y,                         // [TOKS, DINNER] (aliases delta)
    int reverse)
{
  const int idx = blockIdx.x * 256 + threadIdx.x;   // NBATCH*DINNER threads
  const int d   = idx & (DINNER - 1);
  const int bb  = idx >> 11;
  float A[NSTATE], h[NSTATE];
#pragma unroll
  for (int n = 0; n < NSTATE; ++n) {
    A[n] = -__expf(A_log[(size_t)d * NSTATE + n]);
    h[n] = 0.f;
  }
  const float Dv = Dp[d];
  for (int s = 0; s < LSEQ; ++s) {
    const int    t   = reverse ? (LSEQ - 1 - s) : s;
    const size_t tok = (size_t)bb * LSEQ + t;
    const float dlt = (float)delta[tok * DINNER + d];
    const float uu  = (float)u[tok * DINNER + d];
    const float du  = dlt * uu;
    const float* Bt = Bm + tok * NSTATE;   // uniform across lanes -> broadcast
    const float* Ct = Cm + tok * NSTATE;
    float yv = 0.f;
#pragma unroll
    for (int n = 0; n < NSTATE; ++n) {
      const float dA = __expf(dlt * A[n]);
      h[n] = h[n] * dA + du * Bt[n];
      yv += h[n] * Ct[n];
    }
    const float z  = (float)xz[tok * (2 * DINNER) + DINNER + d];
    const float sz = z / (1.f + __expf(-z));
    y[tok * DINNER + d] = (__bf16)((yv + uu * Dv) * sz);
  }
}

// ------------------------------ host side ----------------------------------

extern "C" void kernel_launch(void* const* d_in, const int* in_sizes, int n_in,
                              void* d_out, int out_size, void* d_ws, size_t ws_size,
                              hipStream_t stream)
{
  (void)in_sizes; (void)n_in; (void)out_size; (void)ws_size;

  const float* x     = (const float*)d_in[0];
  const float* ln_g  = (const float*)d_in[1];
  const float* ln_b  = (const float*)d_in[2];
  const float* cmb_w = (const float*)d_in[3];
  const float* cmb_b = (const float*)d_in[4];
  const float* in_w[2]   = {(const float*)d_in[5],  (const float*)d_in[14]};
  const float* conv_w[2] = {(const float*)d_in[6],  (const float*)d_in[15]};
  const float* conv_b[2] = {(const float*)d_in[7],  (const float*)d_in[16]};
  const float* x_w[2]    = {(const float*)d_in[8],  (const float*)d_in[17]};
  const float* dt_w[2]   = {(const float*)d_in[9],  (const float*)d_in[18]};
  const float* dt_b[2]   = {(const float*)d_in[10], (const float*)d_in[19]};
  const float* A_log[2]  = {(const float*)d_in[11], (const float*)d_in[20]};
  const float* Dp[2]     = {(const float*)d_in[12], (const float*)d_in[21]};
  const float* out_w[2]  = {(const float*)d_in[13], (const float*)d_in[22]};

  // --- workspace carve-out (256B aligned) ---
  char*  base = (char*)d_ws;
  size_t off  = 0;
  auto alloc = [&](size_t bytes) -> void* {
    void* p = base + off;
    off = (off + bytes + 255) & ~(size_t)255;
    return p;
  };

  __bf16* xn_bf      = (__bf16*)alloc((size_t)TOKS * DMODEL * 2);
  __bf16* w_in_bf[2], *w_x_bf[2], *w_dt_bf[2], *w_out_bf[2];
  for (int dir = 0; dir < 2; ++dir) {
    w_in_bf[dir]  = (__bf16*)alloc((size_t)2 * DINNER * DMODEL * 2);
    w_x_bf[dir]   = (__bf16*)alloc((size_t)(DTRANK + 2 * NSTATE) * DINNER * 2);
    w_dt_bf[dir]  = (__bf16*)alloc((size_t)DINNER * DTRANK * 2);
    w_out_bf[dir] = (__bf16*)alloc((size_t)DMODEL * DINNER * 2);
  }
  __bf16* w_cmb_bf   = (__bf16*)alloc((size_t)DMODEL * 2 * DMODEL * 2);
  __bf16* xz_bf[2], *u_bf[2], *dt_bf[2], *delta_bf[2];
  float *Bmf[2], *Cmf[2];
  for (int dir = 0; dir < 2; ++dir) {
    xz_bf[dir]    = (__bf16*)alloc((size_t)TOKS * 2 * DINNER * 2);
    u_bf[dir]     = (__bf16*)alloc((size_t)TOKS * DINNER * 2);
    dt_bf[dir]    = (__bf16*)alloc((size_t)TOKS * DTRANK * 2);
    delta_bf[dir] = (__bf16*)alloc((size_t)TOKS * DINNER * 2);   // also y
    Bmf[dir]      = (float*)alloc((size_t)TOKS * NSTATE * 4);
    Cmf[dir]      = (float*)alloc((size_t)TOKS * NSTATE * 4);
  }
  __bf16* ocat_bf = (__bf16*)alloc((size_t)TOKS * 2 * DMODEL * 2);

  const dim3 blk(256);
  auto cvt = [&](const float* s, __bf16* d, int n) {
    f32_to_bf16_kernel<<<dim3((n + 255) / 256), blk, 0, stream>>>(s, d, n);
  };
  auto ggrid = [](int M, int N) { return dim3(M / 128, (N + 127) / 128); };

  // --- weights fp32 -> bf16 ---
  for (int dir = 0; dir < 2; ++dir) {
    cvt(in_w[dir],  w_in_bf[dir],  2 * DINNER * DMODEL);
    cvt(x_w[dir],   w_x_bf[dir],   (DTRANK + 2 * NSTATE) * DINNER);
    cvt(dt_w[dir],  w_dt_bf[dir],  DINNER * DTRANK);
    cvt(out_w[dir], w_out_bf[dir], DMODEL * DINNER);
  }
  cvt(cmb_w, w_cmb_bf, DMODEL * 2 * DMODEL);

  // --- layernorm -> xn (bf16) ---
  layernorm_bf16_kernel<<<dim3(TOKS), blk, 0, stream>>>(x, ln_g, ln_b, xn_bf);

  // --- per-direction Mamba ---
  for (int dir = 0; dir < 2; ++dir) {
    const int rev = dir;   // 0 = forward, 1 = backward (time-space formulation)

    // in-projection: [TOKS,1024] @ [4096,1024]^T -> xz bf16 [TOKS,4096]
    gemm_bf16_wmma<false><<<ggrid(TOKS, 2 * DINNER), blk, 0, stream>>>(
        xn_bf, DMODEL, w_in_bf[dir], DMODEL, 2 * DINNER, DMODEL,
        EpiBF16{xz_bf[dir], 2 * DINNER, 0});

    // depthwise conv + silu -> u bf16 [TOKS,2048]
    conv_silu_kernel<<<dim3((TOKS * DINNER) / 256), blk, 0, stream>>>(
        xz_bf[dir], conv_w[dir], conv_b[dir], u_bf[dir], rev);

    // dbc: [TOKS,2048] @ [96,2048]^T -> dt(bf16) | B(f32) | C(f32)
    gemm_bf16_wmma<true><<<ggrid(TOKS, DTRANK + 2 * NSTATE), blk, 0, stream>>>(
        u_bf[dir], DINNER, w_x_bf[dir], DINNER, DTRANK + 2 * NSTATE, DINNER,
        EpiDbc{dt_bf[dir], Bmf[dir], Cmf[dir]});

    // delta: softplus(dt @ dt_w^T + dt_b) -> bf16 [TOKS,2048]
    gemm_bf16_wmma<false><<<ggrid(TOKS, DINNER), blk, 0, stream>>>(
        dt_bf[dir], DTRANK, w_dt_bf[dir], DTRANK, DINNER, DTRANK,
        EpiDelta{dt_b[dir], delta_bf[dir]});

    // selective scan (+D skip, +silu(z) gate); y overwrites delta in place
    ssm_scan_kernel<<<dim3((NBATCH * DINNER) / 256), blk, 0, stream>>>(
        delta_bf[dir], u_bf[dir], Bmf[dir], Cmf[dir],
        A_log[dir], Dp[dir], xz_bf[dir], delta_bf[dir], rev);

    // out-projection into concat buffer columns [dir*1024, dir*1024+1024)
    gemm_bf16_wmma<false><<<ggrid(TOKS, DMODEL), blk, 0, stream>>>(
        delta_bf[dir], DINNER, w_out_bf[dir], DINNER, DMODEL, DINNER,
        EpiBF16{ocat_bf, 2 * DMODEL, dir * DMODEL});
  }

  // --- combine: [TOKS,2048] @ [1024,2048]^T + bias + residual -> out f32 ---
  gemm_bf16_wmma<false><<<ggrid(TOKS, DMODEL), blk, 0, stream>>>(
      ocat_bf, 2 * DMODEL, w_cmb_bf, 2 * DMODEL, DMODEL, 2 * DMODEL,
      EpiCombine{cmb_b, x, (float*)d_out});
}
